// PartitionedTransformerEncoderLayer_16922171147055
// MI455X (gfx1250) — matrix operand order repaired and checked by reading, not compiled, not run on hardware
//
#include <hip/hip_runtime.h>
#include <hip/hip_bf16.h>

typedef __attribute__((ext_vector_type(16))) __bf16 v16bf;
typedef __attribute__((ext_vector_type(8)))  float  v8f;
typedef __attribute__((ext_vector_type(4)))  unsigned int u32x4;

#define LN_EPS 1e-5f

// ---------------- CDNA5 async-to-LDS helpers (inline asm, ISA 15.18.3) ------
typedef __attribute__((address_space(3))) const void* lds_cvptr;

__device__ inline unsigned lds_u32(const void* p) {
  // generic -> LDS addrspace -> raw LDS byte offset (wave32 LDS addressing)
  return (unsigned)(unsigned long long)(lds_cvptr)p;
}

__device__ inline void async_copy16(unsigned lds, const void* gptr) {
  // 16 bytes per lane, memory -> LDS, tracked by ASYNCcnt
  asm volatile("global_load_async_to_lds_b128 %0, %1, off"
               :: "v"(lds), "v"((unsigned long long)gptr) : "memory");
}
__device__ inline void wait_async_le4() {
  asm volatile("s_wait_asynccnt 0x4" ::: "memory");
}
__device__ inline void wait_async_0() {
  asm volatile("s_wait_asynccnt 0x0" ::: "memory");
}

// Build a 16-element bf16 WMMA fragment from two 16-byte LDS chunks.
__device__ inline v16bf ld_frag16(const __bf16* p0, const __bf16* p1) {
  union { v16bf v; u32x4 q[2]; } u;
  u.q[0] = *(const u32x4*)p0;
  u.q[1] = *(const u32x4*)p1;
  return u.v;
}

__device__ inline v8f wmma_bf16(v16bf a, v16bf b, v8f c) {
  return __builtin_amdgcn_wmma_f32_16x16x32_bf16(false, a, false, b, (short)0, c,
                                                 false, false);
}

// ---------------------------------------------------------------------------
// bf16 WMMA GEMM, async double-buffered staging.
// A: bf16 [M][lda], column offset coff (tile rows contiguous)
// Bt: bf16 [N][ldb] (k contiguous per n-row)
// Block = 64x64 C tile, 4 waves; per-wave 32x32 (2x2 WMMA frags); K step 32.
// ---------------------------------------------------------------------------
template <class EP>
__global__ __launch_bounds__(128) void gemm64_bf16(
    const __bf16* __restrict__ A, int lda, int coff,
    const __bf16* __restrict__ Bt, int ldb, EP ep, int K) {
  __shared__ __bf16 As[2][64][32];
  __shared__ __bf16 Bs[2][64][32];
  const int tid  = threadIdx.x;
  const int wave = tid >> 5, lane = tid & 31;
  const int l15  = lane & 15, kh = lane >> 4;
  const int wm   = (wave >> 1) * 32, wn = (wave & 1) * 32;
  const int m0   = blockIdx.x * 64, n0 = blockIdx.y * 64;

  // per-thread 16B chunk coords: 256 chunks per 64x32 tile, 2 per thread/operand
  const int r0 = tid >> 2, c0 = tid & 3;  // chunk rows 0..31 / col-chunk 0..3
  const int r1 = r0 + 32;                 // chunk rows 32..63
  const __bf16* Ab = A + (size_t)m0 * lda + coff;
  const __bf16* Bb = Bt + (size_t)n0 * ldb;

  auto prefetch = [&](int k0, int buf) {
    unsigned la = lds_u32(&As[buf][0][0]);
    unsigned lb = lds_u32(&Bs[buf][0][0]);
    async_copy16(la + r0 * 64 + c0 * 16, Ab + (size_t)r0 * lda + k0 + c0 * 8);
    async_copy16(la + r1 * 64 + c0 * 16, Ab + (size_t)r1 * lda + k0 + c0 * 8);
    async_copy16(lb + r0 * 64 + c0 * 16, Bb + (size_t)r0 * ldb + k0 + c0 * 8);
    async_copy16(lb + r1 * 64 + c0 * 16, Bb + (size_t)r1 * ldb + k0 + c0 * 8);
  };

  v8f acc[2][2] = {};
  const int T = K >> 5;
  prefetch(0, 0);
  for (int t = 0; t < T; ++t) {
    if (t + 1 < T) { prefetch((t + 1) << 5, (t + 1) & 1); wait_async_le4(); }
    else           { wait_async_0(); }
    __syncthreads();
    const int buf = t & 1;
    #pragma unroll
    for (int mi = 0; mi < 2; ++mi) {
      const __bf16* ar = &As[buf][wm + mi * 16 + l15][0];
      v16bf a = ld_frag16(ar + kh * 8, ar + 16 + kh * 8);
      #pragma unroll
      for (int ni = 0; ni < 2; ++ni) {
        const __bf16* br = &Bs[buf][wn + ni * 16 + l15][0];
        v16bf b = ld_frag16(br + kh * 8, br + 16 + kh * 8);
        acc[mi][ni] = wmma_bf16(a, b, acc[mi][ni]);
      }
    }
    __syncthreads();
  }
  #pragma unroll
  for (int mi = 0; mi < 2; ++mi)
    #pragma unroll
    for (int ni = 0; ni < 2; ++ni)
      #pragma unroll
      for (int r = 0; r < 8; ++r)
        ep(m0 + wm + mi * 16 + kh * 8 + r, n0 + wn + ni * 16 + l15,
           acc[mi][ni][r]);
}

// ---------------- epilogue functors ----------------
struct EPQkv {  // scatter to Q/K/V bf16 [B,H,S,128]; scale Q by 1/sqrt(128)
  __bf16 *q, *k, *v; int half;
  __device__ void operator()(int row, int n, float val) const {
    int h = n / 192, c = (n % 192) >> 6, a = n & 63;
    int b = row >> 10, t = row & 1023;
    size_t idx = (((size_t)(b * 8 + h)) * 1024 + t) * 128 + half * 64 + a;
    if (c == 0)      q[idx] = (__bf16)(val * 0.08838834764831845f);
    else if (c == 1) k[idx] = (__bf16)val;
    else             v[idx] = (__bf16)val;
  }
};
struct EPStore {
  float* y; int ld, coff;
  __device__ void operator()(int row, int n, float val) const {
    y[(size_t)row * ld + coff + n] = val;
  }
};
struct EPBiasReluBf {
  __bf16* y; const float* bias;
  __device__ void operator()(int row, int n, float val) const {
    float t = val + bias[n];
    y[(size_t)row * 2048 + n] = (__bf16)(t > 0.f ? t : 0.f);
  }
};
struct EPBias {
  float* y; int coff; const float* bias;
  __device__ void operator()(int row, int n, float val) const {
    y[(size_t)row * 1024 + coff + n] = val + bias[n];
  }
};

// ---------------- packing kernels (one-pass f32 -> bf16) ----------------
__global__ __launch_bounds__(256) void cvt_bf16(const float* __restrict__ x,
                                                __bf16* __restrict__ y, int n) {
  int i = blockIdx.x * 256 + threadIdx.x;
  if (i < n) y[i] = (__bf16)x[i];
}
// w_o (K=512 rows, N=512 cols, row-major) -> Bt[n][k]
__global__ __launch_bounds__(256) void pack_wo(const float* __restrict__ w,
                                               __bf16* __restrict__ bt) {
  int i = blockIdx.x * 256 + threadIdx.x;   // over N*K = 512*512
  int nn = i >> 9, kk = i & 511;
  bt[i] = (__bf16)w[kk * 512 + nn];
}
// w_qkv (8,512,3,64) -> Bt[n][k], n = h*192 + c*64 + a, K=512
__global__ __launch_bounds__(256) void pack_wqkv(const float* __restrict__ w,
                                                 __bf16* __restrict__ bt) {
  int i = blockIdx.x * 256 + threadIdx.x;   // over 1536*512
  int nn = i >> 9, kk = i & 511;
  int h = nn / 192, c = (nn % 192) >> 6, a = nn & 63;
  bt[i] = (__bf16)w[(((size_t)h * 512 + kk) * 3 + c) * 64 + a];
}

// ---------------------------------------------------------------------------
// Flash attention: block = (b,h, 64-row Q tile), 4 waves; wave owns 16 Q rows.
// K chunk staged with async-to-LDS; V transposed via DS stores; online softmax.
// Writes bf16 A-matrices for output projection: Ac/Ap [B*S][512] (k = h*64+a).
// ---------------------------------------------------------------------------
__global__ __launch_bounds__(128) void flash_attn(
    const __bf16* __restrict__ Qg, const __bf16* __restrict__ Kg,
    const __bf16* __restrict__ Vg, const unsigned char* __restrict__ mask,
    __bf16* __restrict__ Ac, __bf16* __restrict__ Ap) {
  __shared__ __bf16 Ks[64][128];   // keys row-major
  __shared__ __bf16 Vt[128][64];   // values transposed
  __shared__ __bf16 Ps[4][16][64]; // per-wave probability tile

  const int tid = threadIdx.x, wave = tid >> 5, lane = tid & 31;
  const int l15 = lane & 15, kh = lane >> 4;
  const int bh = blockIdx.x, b = bh >> 3, h = bh & 7;
  const int q0 = blockIdx.y * 64 + wave * 16;
  const size_t base = (size_t)bh * 1024 * 128;

  v16bf qf[4];
  {
    const __bf16* qr = Qg + base + (size_t)(q0 + l15) * 128;
    #pragma unroll
    for (int ks = 0; ks < 4; ++ks)
      qf[ks] = ld_frag16(qr + ks * 32 + kh * 8, qr + ks * 32 + 16 + kh * 8);
  }

  v8f o[8] = {};
  float mrun[8], lrun[8];
  #pragma unroll
  for (int r = 0; r < 8; ++r) { mrun[r] = -3.0e38f; lrun[r] = 0.f; }

  const unsigned lk = lds_u32(&Ks[0][0]);

  for (int j0 = 0; j0 < 1024; j0 += 64) {
    // async-stage K chunk (16KB = 8 x b128 per thread)
    {
      const __bf16* ksrc = Kg + base + (size_t)j0 * 128;
      #pragma unroll
      for (int i = tid; i < 1024; i += 128)
        async_copy16(lk + i * 16, ksrc + i * 8);
      // transpose-stage V chunk
      const __bf16* vsrc = Vg + base + (size_t)j0 * 128;
      #pragma unroll
      for (int i = tid; i < 8192; i += 128) Vt[i & 127][i >> 7] = vsrc[i];
    }
    wait_async_0();
    __syncthreads();

    // S = Q @ K^T : 16x64 per wave (Q pre-scaled)
    float sfr[4][8];
    #pragma unroll
    for (int nt = 0; nt < 4; ++nt) {
      v8f acc = {};
      #pragma unroll
      for (int ks = 0; ks < 4; ++ks) {
        const __bf16* br = &Ks[nt * 16 + l15][ks * 32 + kh * 8];
        acc = wmma_bf16(qf[ks], ld_frag16(br, br + 16), acc);
      }
      int j = j0 + nt * 16 + l15;
      float mb = mask[b * 1024 + j] ? 0.f : -1.0e30f;
      #pragma unroll
      for (int r = 0; r < 8; ++r) sfr[nt][r] = acc[r] + mb;
    }

    // online softmax per row
    #pragma unroll
    for (int r = 0; r < 8; ++r) {
      float mx = fmaxf(fmaxf(sfr[0][r], sfr[1][r]), fmaxf(sfr[2][r], sfr[3][r]));
      mx = fmaxf(mx, __shfl_xor(mx, 1, 32));
      mx = fmaxf(mx, __shfl_xor(mx, 2, 32));
      mx = fmaxf(mx, __shfl_xor(mx, 4, 32));
      mx = fmaxf(mx, __shfl_xor(mx, 8, 32));
      float mnew = fmaxf(mrun[r], mx);
      float sc = __expf(mrun[r] - mnew);
      float rs = 0.f;
      #pragma unroll
      for (int nt = 0; nt < 4; ++nt) {
        float p = __expf(sfr[nt][r] - mnew);
        sfr[nt][r] = p;
        rs += p;
      }
      rs += __shfl_xor(rs, 1, 32);
      rs += __shfl_xor(rs, 2, 32);
      rs += __shfl_xor(rs, 4, 32);
      rs += __shfl_xor(rs, 8, 32);
      lrun[r] = lrun[r] * sc + rs;
      mrun[r] = mnew;
      #pragma unroll
      for (int d = 0; d < 8; ++d) o[d][r] *= sc;
    }

    // P (C-frag) -> LDS -> A-frag (wave-private, in-order DS)
    #pragma unroll
    for (int nt = 0; nt < 4; ++nt)
      #pragma unroll
      for (int r = 0; r < 8; ++r)
        Ps[wave][kh * 8 + r][nt * 16 + l15] = (__bf16)sfr[nt][r];

    // O += P @ V
    #pragma unroll
    for (int ks = 0; ks < 2; ++ks) {
      const __bf16* pr = &Ps[wave][l15][ks * 32];
      v16bf a = ld_frag16(pr + kh * 8, pr + 16 + kh * 8);
      #pragma unroll
      for (int d = 0; d < 8; ++d) {
        const __bf16* br = &Vt[d * 16 + l15][ks * 32 + kh * 8];
        o[d] = wmma_bf16(a, ld_frag16(br, br + 16), o[d]);
      }
    }
    __syncthreads();
  }

  // Normalize; write bf16 A-halves for the output projection (k = h*64 + a)
  #pragma unroll
  for (int r = 0; r < 8; ++r) {
    int t = q0 + kh * 8 + r;
    float inv = 1.f / lrun[r];
    size_t rowb = (size_t)(b * 1024 + t) * 512 + h * 64;
    #pragma unroll
    for (int d = 0; d < 8; ++d) {
      float val = o[d][r] * inv;
      int c = d * 16 + l15;
      if (d < 4) Ac[rowb + c]      = (__bf16)val;
      else       Ap[rowb + c - 64] = (__bf16)val;
    }
  }
}

// ---------------------------------------------------------------------------
// y = LayerNorm(xa + xb) * g + b ; optional bf16 copy for downstream GEMMs
// ---------------------------------------------------------------------------
__global__ __launch_bounds__(256) void ln_residual(
    const float* __restrict__ xa, const float* __restrict__ xb,
    const float* __restrict__ g, const float* __restrict__ bt,
    float* __restrict__ y, __bf16* __restrict__ ybf) {
  __shared__ float rs[256], rq[256];
  const int row = blockIdx.x, tid = threadIdx.x;
  const float* pa = xa + (size_t)row * 1024;
  const float* pb = xb + (size_t)row * 1024;
  float v[4], s = 0.f, s2 = 0.f;
  #pragma unroll
  for (int i = 0; i < 4; ++i) {
    int c = tid + i * 256;
    v[i] = pa[c] + pb[c];
    s += v[i];
    s2 += v[i] * v[i];
  }
  rs[tid] = s; rq[tid] = s2;
  __syncthreads();
  for (int o = 128; o > 0; o >>= 1) {
    if (tid < o) { rs[tid] += rs[tid + o]; rq[tid] += rq[tid + o]; }
    __syncthreads();
  }
  float mu  = rs[0] * (1.f / 1024.f);
  float var = rq[0] * (1.f / 1024.f) - mu * mu;
  float inv = rsqrtf(var + LN_EPS);
  float* py = y + (size_t)row * 1024;
  #pragma unroll
  for (int i = 0; i < 4; ++i) {
    int c = tid + i * 256;
    float o = (v[i] - mu) * inv * g[c] + bt[c];
    py[c] = o;
    if (ybf) ybf[(size_t)row * 1024 + c] = (__bf16)o;
  }
}

// ---------------------------------------------------------------------------
extern "C" void kernel_launch(void* const* d_in, const int* in_sizes, int n_in,
                              void* d_out, int out_size, void* d_ws,
                              size_t ws_size, hipStream_t stream) {
  (void)in_sizes; (void)n_in; (void)out_size; (void)ws_size;
  const float* x        = (const float*)d_in[0];
  const unsigned char* mask = (const unsigned char*)d_in[1];
  const float* w_qkv_c  = (const float*)d_in[2];
  const float* w_qkv_p  = (const float*)d_in[3];
  const float* w_o_c    = (const float*)d_in[4];
  const float* w_o_p    = (const float*)d_in[5];
  const float* w1_c     = (const float*)d_in[6];
  const float* b1_c     = (const float*)d_in[7];
  const float* w1_p     = (const float*)d_in[8];
  const float* b1_p     = (const float*)d_in[9];
  const float* w2_c     = (const float*)d_in[10];
  const float* b2_c     = (const float*)d_in[11];
  const float* w2_p     = (const float*)d_in[12];
  const float* b2_p     = (const float*)d_in[13];
  const float* ln1_g    = (const float*)d_in[14];
  const float* ln1_b    = (const float*)d_in[15];
  const float* ln2_g    = (const float*)d_in[16];
  const float* ln2_b    = (const float*)d_in[17];

  char* p = (char*)d_ws;
  auto alloc = [&](size_t bytes) {
    char* r = p;
    p += (bytes + 255) & ~(size_t)255;
    return r;
  };
  __bf16* xbf  = (__bf16*)alloc(4096ull * 1024 * 2);
  __bf16* Qb   = (__bf16*)alloc(4ull * 8 * 1024 * 128 * 2);
  __bf16* Kb   = (__bf16*)alloc(4ull * 8 * 1024 * 128 * 2);
  __bf16* Vb   = (__bf16*)alloc(4ull * 8 * 1024 * 128 * 2);
  __bf16* Acb  = (__bf16*)alloc(4096ull * 512 * 2);
  __bf16* Apb  = (__bf16*)alloc(4096ull * 512 * 2);
  float*  abuf = (float*)alloc(4096ull * 1024 * 4);   // attn out, reused for FFN
  float*  y1   = (float*)alloc(4096ull * 1024 * 4);
  __bf16* y1bf = (__bf16*)alloc(4096ull * 1024 * 2);
  __bf16* hbf  = (__bf16*)alloc(4096ull * 2048 * 2);
  __bf16* wqc  = (__bf16*)alloc(1536ull * 512 * 2);
  __bf16* wqp  = (__bf16*)alloc(1536ull * 512 * 2);
  __bf16* woc  = (__bf16*)alloc(512ull * 512 * 2);
  __bf16* wop  = (__bf16*)alloc(512ull * 512 * 2);
  __bf16* w1c  = (__bf16*)alloc(2048ull * 512 * 2);
  __bf16* w1p  = (__bf16*)alloc(2048ull * 512 * 2);
  __bf16* w2c  = (__bf16*)alloc(512ull * 2048 * 2);
  __bf16* w2p  = (__bf16*)alloc(512ull * 2048 * 2);

  dim3 blk(128);

  // --- one-pass bf16 packing ---
  cvt_bf16<<<16384, 256, 0, stream>>>(x, xbf, 4096 * 1024);
  pack_wqkv<<<3072, 256, 0, stream>>>(w_qkv_c, wqc);
  pack_wqkv<<<3072, 256, 0, stream>>>(w_qkv_p, wqp);
  pack_wo<<<1024, 256, 0, stream>>>(w_o_c, woc);
  pack_wo<<<1024, 256, 0, stream>>>(w_o_p, wop);
  cvt_bf16<<<4096, 256, 0, stream>>>(w1_c, w1c, 2048 * 512);   // already [n][k]
  cvt_bf16<<<4096, 256, 0, stream>>>(w1_p, w1p, 2048 * 512);
  cvt_bf16<<<4096, 256, 0, stream>>>(w2_c, w2c, 512 * 2048);   // already [n][k]
  cvt_bf16<<<4096, 256, 0, stream>>>(w2_p, w2p, 512 * 2048);

  // --- QKV projections (M=4096, N=1536, K=512) ---
  {
    dim3 g(4096 / 64, 1536 / 64);
    gemm64_bf16<<<g, blk, 0, stream>>>(xbf, 1024, 0,   wqc, 512,
                                       EPQkv{Qb, Kb, Vb, 0}, 512);
    gemm64_bf16<<<g, blk, 0, stream>>>(xbf, 1024, 512, wqp, 512,
                                       EPQkv{Qb, Kb, Vb, 1}, 512);
  }

  // --- flash attention -> bf16 A-halves ---
  flash_attn<<<dim3(32, 16), blk, 0, stream>>>(Qb, Kb, Vb, mask, Acb, Apb);

  // --- output projection (M=4096, N=512, K=512) per half ---
  {
    dim3 g(4096 / 64, 512 / 64);
    gemm64_bf16<<<g, blk, 0, stream>>>(Acb, 512, 0, woc, 512,
                                       EPStore{abuf, 1024, 0}, 512);
    gemm64_bf16<<<g, blk, 0, stream>>>(Apb, 512, 0, wop, 512,
                                       EPStore{abuf, 1024, 512}, 512);
  }

  // --- LN1: y1 = LN(x + attn) (+ bf16 copy) ---
  ln_residual<<<4096, 256, 0, stream>>>(x, abuf, ln1_g, ln1_b, y1, y1bf);

  // --- FFN (halves sequential; abuf reused as FFN output) ---
  {
    dim3 g1(4096 / 64, 2048 / 64), g2(4096 / 64, 512 / 64);
    gemm64_bf16<<<g1, blk, 0, stream>>>(y1bf, 1024, 0,   w1c, 512,
                                        EPBiasReluBf{hbf, b1_c}, 512);
    gemm64_bf16<<<g2, blk, 0, stream>>>(hbf, 2048, 0, w2c, 2048,
                                        EPBias{abuf, 0, b2_c}, 2048);
    gemm64_bf16<<<g1, blk, 0, stream>>>(y1bf, 1024, 512, w1p, 512,
                                        EPBiasReluBf{hbf, b1_p}, 512);
    gemm64_bf16<<<g2, blk, 0, stream>>>(hbf, 2048, 0, w2p, 2048,
                                        EPBias{abuf, 512, b2_p}, 2048);
  }

  // --- LN2: out = LN(y1 + ff) ---
  ln_residual<<<4096, 256, 0, stream>>>(y1, abuf, ln2_g, ln2_b,
                                        (float*)d_out, (__bf16*)nullptr);
}
